// DeltaNet_49512382988652
// MI455X (gfx1250) — compile-verified, matrix operand order checked by reading, not run on hardware
//
#include <hip/hip_runtime.h>
#include <hip/hip_bf16.h>
#include <math.h>

#define TSEQ   2048
#define DMODEL 768
#define NLAYER 12
#define NHEAD  12
#define DKH    64
#define FFD    2048
#define VOCAB  50304
#define KCONV  4
#define EPSN   1e-6f

typedef __attribute__((ext_vector_type(16))) __bf16 v16bf;
typedef __attribute__((ext_vector_type(8)))  float  v8f;

// fp32 pair -> packed bf16 dword (RNE). High halves merged with one v_perm_b32.
__device__ __forceinline__ unsigned pack2bf(float a, float b) {
  unsigned ua = __float_as_uint(a), ub = __float_as_uint(b);
  ua += 0x7FFFu + ((ua >> 16) & 1u);
  ub += 0x7FFFu + ((ub >> 16) & 1u);
  // dest bytes [3:2] = ub[3:2], [1:0] = ua[3:2]  (sel: 0..3 -> src1, 4..7 -> src0)
  return __builtin_amdgcn_perm(ub, ua, 0x07060302u);
}

// =====================================================================
// Tiled bf16 WMMA GEMM:  C[M,N] = A[M,K] * B[K,N]  (+ optional residual R)
// Block = 128 threads = 4 waves. Tile: BM=128, BN=64, BK=32.
// Wave w owns 2 M-frags x 4 N-frags -> 8 v_wmma per K-tile.
// Double-buffered LDS: global loads for tile kt+1 issued into registers
// before the WMMAs of tile kt; pack+ds_store_b64 after; 1 barrier/K-tile.
// =====================================================================
#define BM 128
#define BN 64
#define BK 32
#define ASTRU 18   // dwords per LDS row = 36 bf16 (72B rows, 8B aligned)

__global__ __launch_bounds__(128) void gemm_bf16_wmma(
    const float* __restrict__ A, const float* __restrict__ B,
    const float* __restrict__ R, float* __restrict__ C,
    int M, int N, int Kd)
{
  __shared__ unsigned As[2][BM * ASTRU];   // [row][k] packed bf16 pairs
  __shared__ unsigned Bs[2][BN * ASTRU];   // [col][k] packed bf16 pairs (B^T)

  const int tid  = threadIdx.x;
  const int lane = tid & 31;
  const int wid  = tid >> 5;       // 0..3
  const int half = lane >> 4;      // 0/1
  const int l16  = lane & 15;
  const int rowBase = blockIdx.y * BM;
  const int colBase = blockIdx.x * BN;

  // per-thread staging bases (strength-reduced addressing)
  const float* aP = A + (size_t)(rowBase + (tid >> 3)) * Kd + (tid & 7) * 4;
  const float* bP = B + (size_t)((tid >> 6) * 4) * N + colBase + (tid & 63);
  const int asOff = (tid >> 3) * ASTRU + (tid & 7) * 2;   // + i*16*ASTRU
  const int bsOff = (tid & 63) * ASTRU + (tid >> 6) * 2;  // + i*4

  v8f acc[2][4] = {};
  float4 ar[8];
  float  br[16];

  auto loadRegs = [&](int k0) {
    #pragma unroll
    for (int i = 0; i < 8; ++i)
      ar[i] = *(const float4*)(aP + (size_t)k0 + (size_t)i * 16 * Kd);
    #pragma unroll
    for (int i = 0; i < 4; ++i) {
      const float* bp = bP + (size_t)(k0 + 8 * i) * N;
      #pragma unroll
      for (int j = 0; j < 4; ++j) br[i * 4 + j] = bp[(size_t)j * N];
    }
  };
  auto storeTiles = [&](int buf) {
    #pragma unroll
    for (int i = 0; i < 8; ++i) {
      uint2 p; p.x = pack2bf(ar[i].x, ar[i].y); p.y = pack2bf(ar[i].z, ar[i].w);
      *(uint2*)&As[buf][asOff + i * 16 * ASTRU] = p;      // ds_store_b64
    }
    #pragma unroll
    for (int i = 0; i < 4; ++i) {
      uint2 p; p.x = pack2bf(br[i * 4 + 0], br[i * 4 + 1]);
      p.y = pack2bf(br[i * 4 + 2], br[i * 4 + 3]);
      *(uint2*)&Bs[buf][bsOff + i * 4] = p;               // ds_store_b64
    }
  };
  auto compute = [&](int buf) {
    // frag dword j holds K-pair; dword index = (j&3) + (j>=4?8:0) + half*4
    union Frag { unsigned u[8]; v16bf v; };
    Frag af[2];
    #pragma unroll
    for (int mf = 0; mf < 2; ++mf) {
      const unsigned* arow = &As[buf][((wid * 2 + mf) * 16 + l16) * ASTRU];
      #pragma unroll
      for (int j = 0; j < 8; ++j)
        af[mf].u[j] = arow[(j & 3) + ((j >> 2) << 3) + half * 4];
    }
    #pragma unroll
    for (int nf = 0; nf < 4; ++nf) {
      Frag bfr;
      const unsigned* brow = &Bs[buf][(nf * 16 + l16) * ASTRU];
      #pragma unroll
      for (int j = 0; j < 8; ++j)
        bfr.u[j] = brow[(j & 3) + ((j >> 2) << 3) + half * 4];
      acc[0][nf] = __builtin_amdgcn_wmma_f32_16x16x32_bf16(
          false, af[0].v, false, bfr.v, (short)0, acc[0][nf], false, false);
      acc[1][nf] = __builtin_amdgcn_wmma_f32_16x16x32_bf16(
          false, af[1].v, false, bfr.v, (short)0, acc[1][nf], false, false);
    }
  };

  loadRegs(0);
  storeTiles(0);
  __syncthreads();

  const int nk = Kd / BK;
  for (int kt = 0; kt < nk; ++kt) {
    const int cur = kt & 1;
    if (kt + 1 < nk) loadRegs((kt + 1) * BK);          // overlap with WMMAs
    if (kt + 2 < nk)                                   // warm L2 for tile kt+2
      __builtin_prefetch(aP + (size_t)(kt + 2) * BK, 0, 1);
    compute(cur);
    if (kt + 1 < nk) {
      storeTiles(cur ^ 1);
      __syncthreads();                                 // single barrier per tile
    }
  }

  // epilogue: C/D layout: VGPR r -> row r + half*8, col = lane%16
  #pragma unroll
  for (int mf = 0; mf < 2; ++mf) {
    #pragma unroll
    for (int nf = 0; nf < 4; ++nf) {
      #pragma unroll
      for (int r = 0; r < 8; ++r) {
        int row = rowBase + (wid * 2 + mf) * 16 + half * 8 + r;
        int col = colBase + nf * 16 + l16;
        size_t off = (size_t)row * N + col;
        float val = acc[mf][nf][r];
        if (R) val += R[off];
        C[off] = val;
      }
    }
  }
}

// =====================================================================
// Elementwise / norm / scan kernels
// =====================================================================
__global__ void embed_kernel(const int* __restrict__ idx,
                             const float* __restrict__ emb,
                             float* __restrict__ h) {
  int i = blockIdx.x * 256 + threadIdx.x;
  if (i >= TSEQ * DMODEL) return;
  int t = i / DMODEL, d = i - t * DMODEL;
  h[i] = emb[(size_t)idx[t] * DMODEL + d];
}

__global__ __launch_bounds__(256) void rmsnorm_kernel(
    const float* __restrict__ x, const float* __restrict__ w,
    float* __restrict__ y, int n) {
  __shared__ float red[256];
  int t = blockIdx.x;
  const float* xr = x + (size_t)t * n;
  float* yr = y + (size_t)t * n;
  float s = 0.f;
  for (int i = threadIdx.x; i < n; i += 256) { float v = xr[i]; s += v * v; }
  red[threadIdx.x] = s; __syncthreads();
  for (int st = 128; st > 0; st >>= 1) {
    if (threadIdx.x < st) red[threadIdx.x] += red[threadIdx.x + st];
    __syncthreads();
  }
  float scale = rsqrtf(red[0] / n + EPSN);
  for (int i = threadIdx.x; i < n; i += 256) yr[i] = xr[i] * scale * w[i];
}

// mode 0: l2norm (sum, no weight); mode 1: rmsnorm over DK with weight
__global__ __launch_bounds__(64) void headnorm_kernel(
    const float* __restrict__ x, const float* __restrict__ w,
    float* __restrict__ y, int mode) {
  __shared__ float red[64];
  size_t base = (size_t)blockIdx.x * DKH;
  float v = x[base + threadIdx.x];
  red[threadIdx.x] = v * v; __syncthreads();
  for (int st = 32; st > 0; st >>= 1) {
    if (threadIdx.x < st) red[threadIdx.x] += red[threadIdx.x + st];
    __syncthreads();
  }
  float ss = red[0];
  float scale = mode ? rsqrtf(ss / DKH + EPSN) : rsqrtf(ss + EPSN);
  float out = v * scale;
  if (mode) out *= w[threadIdx.x];
  y[base + threadIdx.x] = out;
}

__global__ void conv_silu_kernel(const float* __restrict__ x,
                                 const float* __restrict__ w,
                                 float* __restrict__ y) {
  int i = blockIdx.x * 256 + threadIdx.x;
  if (i >= TSEQ * DMODEL) return;
  int t = i / DMODEL, d = i - t * DMODEL;
  float s = 0.f;
  #pragma unroll
  for (int j = 0; j < KCONV; ++j) {
    int tt = t - (KCONV - 1) + j;
    if (tt >= 0) s += x[(size_t)tt * DMODEL + d] * w[d * KCONV + j];
  }
  y[i] = s / (1.f + __expf(-s));
}

__global__ __launch_bounds__(256) void beta_kernel(
    const float* __restrict__ x, const float* __restrict__ Wb,
    float* __restrict__ beta) {
  __shared__ float red[256];
  int t = blockIdx.x;
  const float* xr = x + (size_t)t * DMODEL;
  float part[NHEAD];
  #pragma unroll
  for (int h = 0; h < NHEAD; ++h) part[h] = 0.f;
  for (int d = threadIdx.x; d < DMODEL; d += 256) {
    float xv = xr[d];
    #pragma unroll
    for (int h = 0; h < NHEAD; ++h) part[h] += xv * Wb[d * NHEAD + h];
  }
  for (int h = 0; h < NHEAD; ++h) {
    red[threadIdx.x] = part[h]; __syncthreads();
    for (int st = 128; st > 0; st >>= 1) {
      if (threadIdx.x < st) red[threadIdx.x] += red[threadIdx.x + st];
      __syncthreads();
    }
    if (threadIdx.x == 0)
      beta[(size_t)t * NHEAD + h] = 1.f / (1.f + __expf(-red[0]));
    __syncthreads();
  }
}

// Serial delta-rule scan: 1 block per head; state column j owned by thread j
__global__ __launch_bounds__(64) void delta_kernel(
    const float* __restrict__ q, const float* __restrict__ k,
    const float* __restrict__ v, const float* __restrict__ beta,
    float* __restrict__ o) {
  __shared__ float S[DKH * DKH];
  __shared__ float qv[DKH], kv[DKH], vv[DKH];
  int h = blockIdx.x, j = threadIdx.x;
  for (int i = j; i < DKH * DKH; i += 64) S[i] = 0.f;
  __syncthreads();
  for (int t = 0; t < TSEQ; ++t) {
    size_t base = ((size_t)t * NHEAD + h) * DKH;
    qv[j] = q[base + j]; kv[j] = k[base + j]; vv[j] = v[base + j];
    __syncthreads();
    float b = beta[(size_t)t * NHEAD + h];
    float ks = 0.f;
    #pragma unroll 8
    for (int i = 0; i < DKH; ++i) ks += kv[i] * S[i * DKH + j];   // S^T k
    float dv = b * (vv[j] - ks);
    float out = 0.f;
    #pragma unroll 8
    for (int i = 0; i < DKH; ++i) {
      float s = S[i * DKH + j] + kv[i] * dv;                      // rank-1 update
      S[i * DKH + j] = s;
      out += qv[i] * s;                                           // o = S^T q
    }
    o[base + j] = out;
    __syncthreads();
  }
}

__global__ void silumul_kernel(const float* __restrict__ g,
                               const float* __restrict__ u,
                               float* __restrict__ y, int n) {
  int i = blockIdx.x * 256 + threadIdx.x;
  if (i >= n) return;
  float gv = g[i];
  y[i] = gv / (1.f + __expf(-gv)) * u[i];
}

// Single-pass online log-sum-exp + NLL (reads logits once)
__global__ __launch_bounds__(256) void loss_kernel(
    const float* __restrict__ logits, const int* __restrict__ tgt,
    float* __restrict__ acc) {
  __shared__ float redm[256], reds[256];
  int t = blockIdx.x;
  const float* row = logits + (size_t)t * VOCAB;
  float m = -3.4e38f, s = 0.f;
  for (int i = threadIdx.x; i < VOCAB; i += 256) {
    float v = row[i];
    float nm = fmaxf(m, v);
    s = s * __expf(m - nm) + __expf(v - nm);
    m = nm;
  }
  redm[threadIdx.x] = m; reds[threadIdx.x] = s; __syncthreads();
  for (int st = 128; st > 0; st >>= 1) {
    if (threadIdx.x < st) {
      float m1 = redm[threadIdx.x], s1 = reds[threadIdx.x];
      float m2 = redm[threadIdx.x + st], s2 = reds[threadIdx.x + st];
      float nm = fmaxf(m1, m2);
      reds[threadIdx.x] = s1 * __expf(m1 - nm) + s2 * __expf(m2 - nm);
      redm[threadIdx.x] = nm;
    }
    __syncthreads();
  }
  if (threadIdx.x == 0) {
    int tg = tgt[t];
    float mask = (tg >= 0) ? 1.f : 0.f;
    int tc = tg < 0 ? 0 : (tg >= VOCAB ? VOCAB - 1 : tg);
    float nll = -(row[tc] - redm[0] - __logf(reds[0]));
    atomicAdd(&acc[0], nll * mask);
    atomicAdd(&acc[1], mask);
  }
}

__global__ void loss_final_kernel(const float* __restrict__ acc,
                                  float* __restrict__ out) {
  if (threadIdx.x == 0) out[0] = acc[0] / fmaxf(acc[1], 1.f);
}

// =====================================================================
extern "C" void kernel_launch(void* const* d_in, const int* in_sizes, int n_in,
                              void* d_out, int out_size, void* d_ws, size_t ws_size,
                              hipStream_t stream) {
  (void)in_sizes; (void)n_in; (void)out_size; (void)ws_size;
  const int*   idx       = (const int*)  d_in[0];
  const int*   targets   = (const int*)  d_in[1];
  const float* embed     = (const float*)d_in[2];
  const float* Wq        = (const float*)d_in[3];
  const float* Wk        = (const float*)d_in[4];
  const float* Wv        = (const float*)d_in[5];
  const float* conv_q    = (const float*)d_in[6];
  const float* conv_k    = (const float*)d_in[7];
  const float* conv_v    = (const float*)d_in[8];
  const float* Wb        = (const float*)d_in[9];
  const float* o_norm_w  = (const float*)d_in[10];
  const float* Wo        = (const float*)d_in[11];
  const float* attn_nw   = (const float*)d_in[12];
  const float* mlp_nw    = (const float*)d_in[13];
  const float* Wgate     = (const float*)d_in[14];
  const float* Wup       = (const float*)d_in[15];
  const float* Wdown     = (const float*)d_in[16];
  const float* final_nw  = (const float*)d_in[17];
  const float* lm_head   = (const float*)d_in[18];

  float* logits = (float*)d_out;

  float* ws = (float*)d_ws;
  size_t off = 0;
  auto F = [&](size_t n) { float* p = ws + off; off += n; return p; };
  const size_t TD = (size_t)TSEQ * DMODEL, TF = (size_t)TSEQ * FFD;
  float* h    = F(TD);
  float* x    = F(TD);
  float* pq   = F(TD); float* pk = F(TD); float* pv = F(TD);
  float* q    = F(TD); float* k  = F(TD); float* v  = F(TD);
  float* beta = F((size_t)TSEQ * NHEAD);
  float* o    = F(TD);
  float* on   = F(TD);
  float* y    = F(TD);
  float* g    = F(TF);
  float* u    = F(TF);
  float* act  = F(TF);
  float* hn   = F(TD);
  float* lacc = F(2);

  dim3 blk(128);
  dim3 gD(DMODEL / BN, TSEQ / BM);   // (12, 16)
  dim3 gF(FFD    / BN, TSEQ / BM);   // (32, 16)
  dim3 gV(VOCAB  / BN, TSEQ / BM);   // (786, 16)
  int eltTD = (int)((TD + 255) / 256);
  int eltTF = (int)((TF + 255) / 256);

  embed_kernel<<<eltTD, 256, 0, stream>>>(idx, embed, h);

  for (int l = 0; l < NLAYER; ++l) {
    const size_t oDD = (size_t)l * DMODEL * DMODEL;
    rmsnorm_kernel<<<TSEQ, 256, 0, stream>>>(h, attn_nw + (size_t)l * DMODEL, x, DMODEL);

    gemm_bf16_wmma<<<gD, blk, 0, stream>>>(x, Wq + oDD, nullptr, pq, TSEQ, DMODEL, DMODEL);
    gemm_bf16_wmma<<<gD, blk, 0, stream>>>(x, Wk + oDD, nullptr, pk, TSEQ, DMODEL, DMODEL);
    gemm_bf16_wmma<<<gD, blk, 0, stream>>>(x, Wv + oDD, nullptr, pv, TSEQ, DMODEL, DMODEL);

    const size_t oCK = (size_t)l * DMODEL * KCONV;
    conv_silu_kernel<<<eltTD, 256, 0, stream>>>(pq, conv_q + oCK, q);
    conv_silu_kernel<<<eltTD, 256, 0, stream>>>(pk, conv_k + oCK, k);
    conv_silu_kernel<<<eltTD, 256, 0, stream>>>(pv, conv_v + oCK, v);

    headnorm_kernel<<<TSEQ * NHEAD, 64, 0, stream>>>(q, nullptr, q, 0);  // l2norm
    headnorm_kernel<<<TSEQ * NHEAD, 64, 0, stream>>>(k, nullptr, k, 0);  // l2norm

    beta_kernel<<<TSEQ, 256, 0, stream>>>(x, Wb + (size_t)l * DMODEL * NHEAD, beta);
    delta_kernel<<<NHEAD, 64, 0, stream>>>(q, k, v, beta, o);
    headnorm_kernel<<<TSEQ * NHEAD, 64, 0, stream>>>(o, o_norm_w + (size_t)l * DKH, on, 1);

    gemm_bf16_wmma<<<gD, blk, 0, stream>>>(on, Wo + oDD, h, h, TSEQ, DMODEL, DMODEL);

    rmsnorm_kernel<<<TSEQ, 256, 0, stream>>>(h, mlp_nw + (size_t)l * DMODEL, y, DMODEL);
    const size_t oDF = (size_t)l * DMODEL * FFD;
    gemm_bf16_wmma<<<gF, blk, 0, stream>>>(y, Wgate + oDF, nullptr, g, TSEQ, FFD, DMODEL);
    gemm_bf16_wmma<<<gF, blk, 0, stream>>>(y, Wup   + oDF, nullptr, u, TSEQ, FFD, DMODEL);
    silumul_kernel<<<eltTF, 256, 0, stream>>>(g, u, act, (int)TF);
    gemm_bf16_wmma<<<gD, blk, 0, stream>>>(act, Wdown + (size_t)l * FFD * DMODEL,
                                           h, h, TSEQ, DMODEL, FFD);
  }

  rmsnorm_kernel<<<TSEQ, 256, 0, stream>>>(h, final_nw, hn, DMODEL);
  gemm_bf16_wmma<<<gV, blk, 0, stream>>>(hn, lm_head, nullptr, logits, TSEQ, VOCAB, DMODEL);

  hipMemsetAsync(lacc, 0, 2 * sizeof(float), stream);
  loss_kernel<<<TSEQ, 256, 0, stream>>>(logits, targets, lacc);
  loss_final_kernel<<<1, 32, 0, stream>>>(lacc, logits + (size_t)TSEQ * VOCAB);
}
